// MultiHeadAttention_23149873725840
// MI455X (gfx1250) — matrix operand order repaired and checked by reading, not compiled, not run on hardware
//
#include <hip/hip_runtime.h>
#include <hip/hip_bf16.h>

// MI455X (gfx1250) multi-head attention forward.
// Strategy: f32 inputs -> f16 staged operands (f32 WMMA accumulation),
// v_wmma_f32_16x16x32_f16 for all four GEMMs + both attention matmuls,
// two-pass flash softmax so the 1.6GB attn tensor is written exactly once,
// and CDNA5 async global->LDS (ASYNCcnt) staging for the attention tiles.

#define B_SZ 2
#define S_LEN 4096
#define D_MODEL 768
#define N_HEADS 12
#define DH 64

typedef _Float16 v16h __attribute__((ext_vector_type(16)));
typedef float v8f __attribute__((ext_vector_type(8)));
typedef unsigned int v8u __attribute__((ext_vector_type(8)));

__device__ __forceinline__ v8f wmma_f16(v16h a, v16h b, v8f c) {
  // 8 args: (neg_a, A, neg_b, B, c_mod, C, reuse_a, reuse_b)
  return __builtin_amdgcn_wmma_f32_16x16x32_f16(false, a, false, b, (short)0, c,
                                                false, false);
}

__device__ __forceinline__ v8f zero8() {
  v8f z;
#pragma unroll
  for (int i = 0; i < 8; ++i) z[i] = 0.0f;
  return z;
}

// Async copy of 16 bytes global -> LDS. On AMDGPU the generic address of a
// __shared__ object carries the LDS byte offset in its low 32 bits
// (ISA 10.2: LDS aperture).
__device__ __forceinline__ void async_g2l_b128(void* lds_ptr, const void* gptr) {
  const unsigned lds = (unsigned)(unsigned long long)lds_ptr;
  asm volatile("global_load_async_to_lds_b128 %0, %1, off"
               :
               : "v"(lds), "v"(gptr)
               : "memory");
}

__device__ __forceinline__ void wait_async0() {
  asm volatile("s_wait_asynccnt 0x0" ::: "memory");
}

// A fragment 16x32 f16 from LDS row-major [m][k] (stride in halves).
// Contiguous k order used here (not AMD's table order): lane = m + 16*k[4]; VGPR i holds K = 2i, 2i+1 (of the lane's 16-K half).
__device__ __forceinline__ v16h load_a_frag(const _Float16* p, int stride, int k0, int lane) {
  const int m = lane & 15;
  const int kg = lane >> 4;
  const _Float16* base = p + m * stride + k0 + kg * 16;
  v8u t;
#pragma unroll
  for (int i = 0; i < 8; ++i) {
    const int off = i << 1;
    t[i] = *(const unsigned int*)(base + off);
  }
  return __builtin_bit_cast(v16h, t);
}

// B fragment 32x16 f16 from *transposed* LDS [n][k].
// Contiguous k order used here (not AMD's table order): lane = n + 16*k[4]; VGPR j holds K = 2j, 2j+1 (of the lane's 16-K half).
__device__ __forceinline__ v16h load_b_fragT(const _Float16* p, int stride, int k0, int lane) {
  const int n = lane & 15;
  const int kg = lane >> 4;
  const _Float16* base = p + n * stride + k0 + kg * 16;
  v8u t;
#pragma unroll
  for (int j = 0; j < 8; ++j) t[j] = *(const unsigned int*)(base + 2 * j);
  return __builtin_bit_cast(v16h, t);
}

// B fragment 32x16 f16 from row-major LDS [k][n] (used for V in PV matmul).
__device__ __forceinline__ v16h load_b_fragRM(const _Float16* p, int stride, int k0,
                                              int nBase, int lane) {
  const int n = nBase + (lane & 15);
  const int kg = lane >> 4;
  v16h r;
#pragma unroll
  for (int j = 0; j < 8; ++j) {
    const int k = k0 + kg * 16 + 2 * j;
    r[2 * j] = p[k * stride + n];
    r[2 * j + 1] = p[(k + 1) * stride + n];
  }
  return r;
}

// ---------------------------------------------------------------------------
// QKV projection: X(8192x768) f32 @ W(768x768) f32 + bias -> dst f16 laid out
// as (B, H, S, 64). outScale folds the 1/sqrt(d) attention scale into Q.
// Block = 256 threads (8 waves), tile 128(M) x 64(N), K-step 32.
// ---------------------------------------------------------------------------
__global__ __launch_bounds__(256) void qkv_proj_kernel(
    const float* __restrict__ X, const float* __restrict__ W,
    const float* __restrict__ bias, _Float16* __restrict__ dst, float outScale) {
  __shared__ _Float16 sA[128][36];  // [m][k], pad 4
  __shared__ _Float16 sB[64][36];   // [n][k] transposed W tile
  const int tid = threadIdx.x;
  const int lane = tid & 31;
  const int w = tid >> 5;
  const int mBlk = blockIdx.x;  // 64 blocks over M=8192
  const int h = blockIdx.y;     // 12 N-blocks of 64 == head index

  v8f acc[4];
#pragma unroll
  for (int i = 0; i < 4; ++i) acc[i] = zero8();

  for (int kb = 0; kb < D_MODEL / 32; ++kb) {
    __syncthreads();
    // stage X tile 128x32 (f32 -> f16)
#pragma unroll
    for (int i = 0; i < 4; ++i) {
      const int e = (tid + i * 256) * 4;
      const int r = e >> 5, c = e & 31;
      const float4 v4 =
          *(const float4*)(X + (size_t)(mBlk * 128 + r) * D_MODEL + kb * 32 + c);
      sA[r][c] = (_Float16)v4.x;
      sA[r][c + 1] = (_Float16)v4.y;
      sA[r][c + 2] = (_Float16)v4.z;
      sA[r][c + 3] = (_Float16)v4.w;
    }
    // stage W tile 32x64, transposed to [n][k]
#pragma unroll
    for (int i = 0; i < 2; ++i) {
      const int e = (tid + i * 256) * 4;
      const int r = e >> 6, c = e & 63;
      const float4 v4 =
          *(const float4*)(W + (size_t)(kb * 32 + r) * D_MODEL + h * 64 + c);
      sB[c][r] = (_Float16)v4.x;
      sB[c + 1][r] = (_Float16)v4.y;
      sB[c + 2][r] = (_Float16)v4.z;
      sB[c + 3][r] = (_Float16)v4.w;
    }
    __syncthreads();
    const v16h a = load_a_frag(&sA[w * 16][0], 36, 0, lane);
#pragma unroll
    for (int nt = 0; nt < 4; ++nt) {
      const v16h b = load_b_fragT(&sB[nt * 16][0], 36, 0, lane);
      acc[nt] = wmma_f16(a, b, acc[nt]);
    }
  }

  // Epilogue: bias + scale, scatter to (B,H,S,64) f16.
#pragma unroll
  for (int nt = 0; nt < 4; ++nt) {
    const int d = nt * 16 + (lane & 15);
    const float bval = bias[h * 64 + d];
#pragma unroll
    for (int r = 0; r < 8; ++r) {
      const int mrow = mBlk * 128 + w * 16 + (lane >> 4) * 8 + r;
      const int bb = mrow >> 12;
      const int ss = mrow & (S_LEN - 1);
      const float val = (acc[nt][r] + bval) * outScale;
      dst[(((size_t)(bb * N_HEADS + h)) * S_LEN + ss) * DH + d] = (_Float16)val;
    }
  }
}

// ---------------------------------------------------------------------------
// Fused attention: per block = 128 q-rows of one (b,h). Two passes over K:
// pass 1 builds per-row (max, sumexp) online; pass 2 recomputes logits,
// writes normalized probs to d_out (the attn output) and accumulates P@V.
// Each wave exclusively owns 16 q-rows -> softmax stats reduce with
// intra-16-lane shfl_xor only. sK (row-major [k][d]) *is* the transposed-B
// operand for Q@K^T. Tiles are staged with async global->LDS b128 (ASYNCcnt).
// ---------------------------------------------------------------------------
__global__ __launch_bounds__(256) void attn_kernel(
    const _Float16* __restrict__ qh, const _Float16* __restrict__ kh,
    const _Float16* __restrict__ vh, const float* __restrict__ mask,
    float* __restrict__ attn, _Float16* __restrict__ ctx) {
  __shared__ _Float16 sQ[128][72];     // 18432 B
  __shared__ _Float16 sK[64][72];      //  9216 B
  __shared__ _Float16 sV[64][72];      //  9216 B
  __shared__ _Float16 sP[8][16][72];   // 18432 B  (per-wave P restage)

  const int tid = threadIdx.x, lane = tid & 31, w = tid >> 5;
  const int qBlk = blockIdx.x;  // 32
  const int bh = blockIdx.y;    // 24
  const int b = bh / N_HEADS, h = bh % N_HEADS;

  // Stage the 128x64 Q block once (already f16, already pre-scaled).
  const _Float16* qbase = qh + ((size_t)bh * S_LEN + qBlk * 128) * DH;
#pragma unroll
  for (int i = 0; i < 4; ++i) {
    const int e = (tid + i * 256) * 8;  // 8 halves = 16B per lane
    const int r = e >> 6, c = e & 63;
    async_g2l_b128(&sQ[r][c], qbase + (size_t)r * DH + c);
  }

  float mI[8], lI[8];
#pragma unroll
  for (int r = 0; r < 8; ++r) {
    mI[r] = -1e30f;
    lI[r] = 0.0f;
  }
  const float* mrow = mask + (size_t)b * S_LEN;

  // ---------------- Pass 1: row max + sumexp ----------------
  for (int kt = 0; kt < S_LEN / 64; ++kt) {
    __syncthreads();
    const _Float16* kbase = kh + ((size_t)bh * S_LEN + kt * 64) * DH;
#pragma unroll
    for (int i = 0; i < 2; ++i) {
      const int e = (tid + i * 256) * 8;
      const int r = e >> 6, c = e & 63;
      async_g2l_b128(&sK[r][c], kbase + (size_t)r * DH + c);
    }
    wait_async0();
    __syncthreads();
    const v16h a0 = load_a_frag(&sQ[w * 16][0], 72, 0, lane);
    const v16h a1 = load_a_frag(&sQ[w * 16][0], 72, 32, lane);
#pragma unroll
    for (int nt = 0; nt < 4; ++nt) {
      v8f acc = zero8();
      acc = wmma_f16(a0, load_b_fragT(&sK[nt * 16][0], 72, 0, lane), acc);
      acc = wmma_f16(a1, load_b_fragT(&sK[nt * 16][0], 72, 32, lane), acc);
      const int col = kt * 64 + nt * 16 + (lane & 15);
      const float mk = mrow[col] * -1.0e9f;
#pragma unroll
      for (int r = 0; r < 8; ++r) {
        const float x = acc[r] + mk;
        const float mo = mI[r];
        const float mn = fmaxf(mo, x);
        lI[r] = lI[r] * __expf(mo - mn) + __expf(x - mn);
        mI[r] = mn;
      }
    }
  }
  // merge across the 16 lanes of each half (they share the same 8 rows)
#pragma unroll
  for (int off = 1; off < 16; off <<= 1) {
#pragma unroll
    for (int r = 0; r < 8; ++r) {
      const float mo = __shfl_xor(mI[r], off);
      const float lo = __shfl_xor(lI[r], off);
      const float mn = fmaxf(mI[r], mo);
      lI[r] = lI[r] * __expf(mI[r] - mn) + lo * __expf(mo - mn);
      mI[r] = mn;
    }
  }
  float rinv[8];
#pragma unroll
  for (int r = 0; r < 8; ++r) rinv[r] = 1.0f / lI[r];

  v8f cacc[4];
#pragma unroll
  for (int i = 0; i < 4; ++i) cacc[i] = zero8();

  // ---------------- Pass 2: write probs + P@V ----------------
  for (int kt = 0; kt < S_LEN / 64; ++kt) {
    __syncthreads();
    const _Float16* kbase = kh + ((size_t)bh * S_LEN + kt * 64) * DH;
    const _Float16* vbase = vh + ((size_t)bh * S_LEN + kt * 64) * DH;
#pragma unroll
    for (int i = 0; i < 2; ++i) {
      const int e = (tid + i * 256) * 8;
      const int r = e >> 6, c = e & 63;
      async_g2l_b128(&sK[r][c], kbase + (size_t)r * DH + c);
      async_g2l_b128(&sV[r][c], vbase + (size_t)r * DH + c);
    }
    wait_async0();
    __syncthreads();
    const v16h a0 = load_a_frag(&sQ[w * 16][0], 72, 0, lane);
    const v16h a1 = load_a_frag(&sQ[w * 16][0], 72, 32, lane);
#pragma unroll
    for (int nt = 0; nt < 4; ++nt) {
      v8f acc = zero8();
      acc = wmma_f16(a0, load_b_fragT(&sK[nt * 16][0], 72, 0, lane), acc);
      acc = wmma_f16(a1, load_b_fragT(&sK[nt * 16][0], 72, 32, lane), acc);
      const int col = kt * 64 + nt * 16 + (lane & 15);
      const float mk = mrow[col] * -1.0e9f;
#pragma unroll
      for (int r = 0; r < 8; ++r) {
        const int lm = (lane >> 4) * 8 + r;
        const float x = acc[r] + mk;
        const float p = __expf(x - mI[r]) * rinv[r];
        const size_t qrow = (size_t)qBlk * 128 + w * 16 + lm;
        attn[((size_t)bh * S_LEN + qrow) * S_LEN + col] = p;
        sP[w][lm][nt * 16 + (lane & 15)] = (_Float16)p;
      }
    }
    __syncthreads();  // sP visible for A-frag re-layout, sV ready
    const v16h p0 = load_a_frag(&sP[w][0][0], 72, 0, lane);
    const v16h p1 = load_a_frag(&sP[w][0][0], 72, 32, lane);
#pragma unroll
    for (int dt = 0; dt < 4; ++dt) {
      cacc[dt] = wmma_f16(p0, load_b_fragRM(&sV[0][0], 72, 0, dt * 16, lane), cacc[dt]);
      cacc[dt] = wmma_f16(p1, load_b_fragRM(&sV[0][0], 72, 32, dt * 16, lane), cacc[dt]);
    }
  }

  // context -> (b, s, h*64+d) f16, ready as A-operand of the output GEMM
#pragma unroll
  for (int dt = 0; dt < 4; ++dt) {
#pragma unroll
    for (int r = 0; r < 8; ++r) {
      const int lm = (lane >> 4) * 8 + r;
      const size_t srow = (size_t)qBlk * 128 + w * 16 + lm;
      const int d = dt * 16 + (lane & 15);
      ctx[((size_t)b * S_LEN + srow) * D_MODEL + h * DH + d] = (_Float16)cacc[dt][r];
    }
  }
}

// ---------------------------------------------------------------------------
// Output projection: ctx(8192x768) f16 @ wo(768x768) f32 + bo -> out f32.
// ---------------------------------------------------------------------------
__global__ __launch_bounds__(256) void out_proj_kernel(
    const _Float16* __restrict__ A, const float* __restrict__ W,
    const float* __restrict__ bias, float* __restrict__ out) {
  __shared__ _Float16 sA[128][36];
  __shared__ _Float16 sB[64][36];
  const int tid = threadIdx.x, lane = tid & 31, w = tid >> 5;
  const int mBlk = blockIdx.x, nBlk = blockIdx.y;

  v8f acc[4];
#pragma unroll
  for (int i = 0; i < 4; ++i) acc[i] = zero8();

  for (int kb = 0; kb < D_MODEL / 32; ++kb) {
    __syncthreads();
#pragma unroll
    for (int i = 0; i < 4; ++i) {
      const int e = (tid + i * 256) * 4;
      const int r = e >> 5, c = e & 31;
      *(uint2*)&sA[r][c] =
          *(const uint2*)(A + (size_t)(mBlk * 128 + r) * D_MODEL + kb * 32 + c);
    }
#pragma unroll
    for (int i = 0; i < 2; ++i) {
      const int e = (tid + i * 256) * 4;
      const int r = e >> 6, c = e & 63;
      const float4 v4 =
          *(const float4*)(W + (size_t)(kb * 32 + r) * D_MODEL + nBlk * 64 + c);
      sB[c][r] = (_Float16)v4.x;
      sB[c + 1][r] = (_Float16)v4.y;
      sB[c + 2][r] = (_Float16)v4.z;
      sB[c + 3][r] = (_Float16)v4.w;
    }
    __syncthreads();
    const v16h a = load_a_frag(&sA[w * 16][0], 36, 0, lane);
#pragma unroll
    for (int nt = 0; nt < 4; ++nt) {
      acc[nt] = wmma_f16(a, load_b_fragT(&sB[nt * 16][0], 36, 0, lane), acc[nt]);
    }
  }

#pragma unroll
  for (int nt = 0; nt < 4; ++nt) {
    const int n = nBlk * 64 + nt * 16 + (lane & 15);
    const float bval = bias[n];
#pragma unroll
    for (int r = 0; r < 8; ++r) {
      const int mrow = mBlk * 128 + w * 16 + (lane >> 4) * 8 + r;
      out[(size_t)mrow * D_MODEL + n] = acc[nt][r] + bval;
    }
  }
}

extern "C" void kernel_launch(void* const* d_in, const int* in_sizes, int n_in,
                              void* d_out, int out_size, void* d_ws, size_t ws_size,
                              hipStream_t stream) {
  (void)in_sizes; (void)n_in; (void)out_size; (void)ws_size;
  const float* v = (const float*)d_in[0];
  const float* k = (const float*)d_in[1];
  const float* q = (const float*)d_in[2];
  const float* mask = (const float*)d_in[3];
  const float* wq = (const float*)d_in[4];
  const float* bq = (const float*)d_in[5];
  const float* wk = (const float*)d_in[6];
  const float* bk = (const float*)d_in[7];
  const float* wv = (const float*)d_in[8];
  const float* bv = (const float*)d_in[9];
  const float* wo = (const float*)d_in[10];
  const float* bo = (const float*)d_in[11];

  const size_t perMat = (size_t)B_SZ * N_HEADS * S_LEN * DH;  // 6291456 halves
  _Float16* qh = (_Float16*)d_ws;
  _Float16* kh = qh + perMat;
  _Float16* vhh = kh + perMat;
  _Float16* ctx = vhh + perMat;  // total ws use: 4*perMat*2 B = 50.3 MB

  float* out = (float*)d_out;
  float* attn = out + (size_t)B_SZ * S_LEN * D_MODEL;

  const dim3 blk(256);
  const dim3 gProj(64, 12);
  // Q projection carries the 1/sqrt(64) softmax scale.
  qkv_proj_kernel<<<gProj, blk, 0, stream>>>(q, wq, bq, qh, 0.125f);
  qkv_proj_kernel<<<gProj, blk, 0, stream>>>(k, wk, bk, kh, 1.0f);
  qkv_proj_kernel<<<gProj, blk, 0, stream>>>(v, wv, bv, vhh, 1.0f);
  attn_kernel<<<dim3(32, 24), blk, 0, stream>>>(qh, kh, vhh, mask, attn, ctx);
  out_proj_kernel<<<gProj, blk, 0, stream>>>(ctx, wo, bo, out);
}
